// GumbelLayer_50422916055590
// MI455X (gfx1250) — compile-verified, matrix operand order checked by reading, not run on hardware
//
#include <hip/hip_runtime.h>

// ---- vector types (raw ext_vector to stay union/bitcast friendly) ----
typedef __attribute__((ext_vector_type(16))) __bf16         v16bf;
typedef __attribute__((ext_vector_type(8)))  float          v8f;
typedef __attribute__((ext_vector_type(4)))  float          f32x4;
typedef __attribute__((ext_vector_type(4)))  unsigned int   u32x4;
typedef __attribute__((ext_vector_type(4)))  unsigned short u16x4;

union FragBF { u32x4 q[2]; v16bf v; };

// fp32 -> bf16 round-to-nearest-even
__device__ __forceinline__ unsigned short f2bf(float f) {
  unsigned u = __builtin_bit_cast(unsigned, f);
  unsigned r = (u + 0x7FFFu + ((u >> 16) & 1u)) >> 16;
  return (unsigned short)r;
}

#define BM   128
#define BN   128
#define BK   32
#define LDSS 40   // padded LDS row stride (bf16 elems): 80B rows, conflict-free b128 reads

// ---- staging split into load phase (global->regs) and store phase (regs->LDS)
// so the global_load_b128 batch can be hoisted above the WMMA work and its
// s_wait_loadcnt lands after ~8 WMMAs of latency hiding. ----
template <bool GUARD>
__device__ __forceinline__ void load_tile(f32x4 (&regs)[4],
                                          const float* __restrict__ gbase, int ldg,
                                          int rowOff, int maxRow, int tid) {
#pragma unroll
  for (int i = 0; i < 4; ++i) {
    int id = tid + i * 256;        // 1024 float4-chunks per tile
    int r  = id >> 3;              // 8 chunks of 4 floats per 32-wide row
    int c4 = (id & 7) << 2;
    int gr = rowOff + r;
    if (GUARD) {
      f32x4 z = {0.0f, 0.0f, 0.0f, 0.0f};
      regs[i] = (gr < maxRow) ? *(const f32x4*)(gbase + (size_t)gr * ldg + c4) : z;
    } else {
      regs[i] = *(const f32x4*)(gbase + (size_t)gr * ldg + c4);
    }
  }
}

__device__ __forceinline__ void store_tile(const f32x4 (&regs)[4],
                                           unsigned short* __restrict__ lds, int tid) {
#pragma unroll
  for (int i = 0; i < 4; ++i) {
    int id = tid + i * 256;
    int r  = id >> 3;
    int c4 = (id & 7) << 2;
    u16x4 h;
    h[0] = f2bf(regs[i][0]); h[1] = f2bf(regs[i][1]);
    h[2] = f2bf(regs[i][2]); h[3] = f2bf(regs[i][3]);
    *(u16x4*)&lds[r * LDSS + c4] = h;   // ds_store_b64, 8B aligned
  }
}

template <bool GUARD>
__global__ __launch_bounds__(256)
void gumbel_wmma_bf16_kernel(const float* __restrict__ x,
                             const float* __restrict__ u1,
                             const float* __restrict__ u2,
                             const float* __restrict__ W,
                             const float* __restrict__ bias,
                             float* __restrict__ out,
                             int Brows, int K, int N) {
  __shared__ unsigned short As[2][BM * LDSS];
  __shared__ unsigned short Bs[2][BN * LDSS];

  const int tid   = threadIdx.x;
  const int lane  = tid & 31;
  const int wid   = tid >> 5;          // 8 waves: 4 (M) x 2 (N)
  const int waveM = (wid & 3) * 32;
  const int waveN = (wid >> 2) * 64;
  const int mrow  = lane & 15;         // M row (A) / N col (B) within 16x16 tile
  const int khalf = lane >> 4;         // selects K-chunk per ISA 16-bit layout

  const int tileM = blockIdx.y * BM;   // rows of x / out
  const int tileN = blockIdx.x * BN;   // cols of out (rows of W)

  v8f acc[2][4] = {};

  const int nk = K / BK;

  // prologue: stage first K-slab into buffer 0
  {
    f32x4 ra[4], rb[4];
    load_tile<GUARD>(ra, x, K, tileM, Brows, tid);
    load_tile<false>(rb, W, K, tileN, N,     tid);
    store_tile(ra, As[0], tid);
    store_tile(rb, Bs[0], tid);
  }

  for (int kt = 0; kt < nk; ++kt) {
    __syncthreads();
    const int cur = kt & 1;
    const bool more = (kt + 1 < nk);

    // (1) issue next slab's global loads immediately (latency hidden by (2))
    f32x4 ra[4], rb[4];
    if (more) {
      const float* xk = x + (kt + 1) * BK;
      const float* wk = W + (kt + 1) * BK;
      load_tile<GUARD>(ra, xk, K, tileM, Brows, tid);
      load_tile<false>(rb, wk, K, tileN, N,     tid);
    }

    // (2) fragment gathers + WMMA on current buffer
    // (ISA 7.12.2: lane holds row mrow, two 8-elem K chunks at k = khalf*8, 16+khalf*8)
    FragBF a[2], bfr[4];
#pragma unroll
    for (int mi = 0; mi < 2; ++mi) {
      const unsigned short* p =
          &As[cur][(waveM + mi * 16 + mrow) * LDSS + khalf * 8];
      a[mi].q[0] = *(const u32x4*)p;          // ds_load_b128
      a[mi].q[1] = *(const u32x4*)(p + 16);
    }
#pragma unroll
    for (int ni = 0; ni < 4; ++ni) {
      const unsigned short* p =
          &Bs[cur][(waveN + ni * 16 + mrow) * LDSS + khalf * 8];
      bfr[ni].q[0] = *(const u32x4*)p;
      bfr[ni].q[1] = *(const u32x4*)(p + 16);
    }
#pragma unroll
    for (int mi = 0; mi < 2; ++mi)
#pragma unroll
      for (int ni = 0; ni < 4; ++ni)
        acc[mi][ni] = __builtin_amdgcn_wmma_f32_16x16x32_bf16(
            false, a[mi].v, false, bfr[ni].v,
            (short)0, acc[mi][ni], false, false);

    // (3) convert + park next slab in the other LDS buffer
    if (more) {
      store_tile(ra, As[cur ^ 1], tid);
      store_tile(rb, Bs[cur ^ 1], tid);
    }
  }

  // ---- fused epilogue: bias + gumbel noise + sigmoid(logit/T), T = 0.1 ----
  // C/D layout: VGPR r holds M = r (lanes 0-15) / M = r + 8 (lanes 16-31), N = lane&15.
  // u1/u2/out are streamed exactly once -> non-temporal (keep L2 for x/W reuse).
#pragma unroll
  for (int mi = 0; mi < 2; ++mi) {
    const int rowB = tileM + waveM + mi * 16 + khalf * 8;
#pragma unroll
    for (int ni = 0; ni < 4; ++ni) {
      const int col = tileN + waveN + ni * 16 + mrow;
      const float bb = bias[col];
#pragma unroll
      for (int r = 0; r < 8; ++r) {
        const int row = rowB + r;
        if (!GUARD || row < Brows) {
          const size_t idx = (size_t)row * N + col;
          const float a1 = __builtin_nontemporal_load(&u1[idx]);
          const float a2 = __builtin_nontemporal_load(&u2[idx]);
          const float l  = acc[mi][ni][r] + bb;
          const float g1 = -__logf(-__logf(a1));
          const float g2 = -__logf(-__logf(a2));
          const float z  = (l + g1 - g2) * 10.0f;   // divide by T=0.1
          const float res = 1.0f / (1.0f + __expf(-z));
          __builtin_nontemporal_store(res, &out[idx]);
        }
      }
    }
  }
}

extern "C" void kernel_launch(void* const* d_in, const int* in_sizes, int n_in,
                              void* d_out, int out_size, void* d_ws, size_t ws_size,
                              hipStream_t stream) {
  (void)n_in; (void)out_size; (void)d_ws; (void)ws_size;
  const float* x  = (const float*)d_in[0];
  const float* u1 = (const float*)d_in[1];
  const float* u2 = (const float*)d_in[2];
  const float* W  = (const float*)d_in[3];
  const float* b  = (const float*)d_in[4];
  // d_in[5] = with_grad (unused by forward)

  const int DOUT  = in_sizes[4];              // 1024
  const int DIN   = in_sizes[3] / DOUT;       // 1024
  const int Brows = in_sizes[0] / DIN;        // 16384

  dim3 grid((DOUT + BN - 1) / BN,             // N fastest -> x-tile reuse in L2
            (Brows + BM - 1) / BM);
  dim3 block(256);

  const bool exact = (Brows % BM == 0) && (DOUT % BN == 0) && (DIN % BK == 0);
  if (exact) {
    gumbel_wmma_bf16_kernel<false><<<grid, block, 0, stream>>>(
        x, u1, u2, W, b, (float*)d_out, Brows, DIN, DOUT);
  } else {
    gumbel_wmma_bf16_kernel<true><<<grid, block, 0, stream>>>(
        x, u1, u2, W, b, (float*)d_out, Brows, DIN, DOUT);
  }
}
